// SS2D_46342697124179
// MI455X (gfx1250) — compile-verified, matrix operand order checked by reading, not run on hardware
//
#include <hip/hip_runtime.h>
#include <hip/hip_bf16.h>

// Problem constants (from reference)
#define BQ   8
#define DM   96
#define DIN  192
#define NST  16
#define RNK  6
#define KD   4
#define LL   4096
#define MTOK (BQ*LL)         // 32768 tokens
#define CDBL (RNK + 2*NST)   // 38

typedef __attribute__((ext_vector_type(16))) __bf16 v16bf;
typedef __attribute__((ext_vector_type(8)))  float  v8f;

// cross_scan / cross_merge permutation (involution per direction)
__device__ __forceinline__ int perm_pos(int k, int l) {
  switch (k & 3) {
    case 0:  return l;
    case 1:  return ((l & 63) << 6) | (l >> 6);
    case 2:  return (LL - 1) - l;
    default: { int j = (LL - 1) - l; return ((j & 63) << 6) | (j >> 6); }
  }
}

// ---------------- pack / init ----------------
__global__ void pack_bf16(const float* __restrict__ src, __bf16* __restrict__ dst, int n) {
  for (int i = blockIdx.x * blockDim.x + threadIdx.x; i < n; i += gridDim.x * blockDim.x)
    dst[i] = (__bf16)src[i];
}

__global__ void zero_f32(float* __restrict__ p, int n) {
  for (int i = blockIdx.x * blockDim.x + threadIdx.x; i < n; i += gridDim.x * blockDim.x)
    p[i] = 0.0f;
}

// ---- WMMA 16x16 tile from LDS-resident row-major tiles (leading dim KDIM) ----
template <int KDIM>
__device__ __forceinline__ v8f wmma_tile(const __bf16* A, const __bf16* B, int lane) {
  const int row = lane & 15;          // M row for A frag / N col for B frag
  const int kb  = (lane >> 4) * 8;    // k-half select (ISA 16-bit 16x32 layout)
  v8f c;
#pragma unroll
  for (int i = 0; i < 8; i++) c[i] = 0.0f;
#pragma unroll
  for (int k0 = 0; k0 < KDIM; k0 += 32) {
    v16bf a, b;
    const __bf16* ap = A + row * KDIM + k0;
    const __bf16* bp = B + row * KDIM + k0;
#pragma unroll
    for (int i = 0; i < 16; i++) {
      int kk = ((i >> 3) << 4) + kb + (i & 7);
      a[i] = ap[kk];
      b[i] = bp[kk];
    }
    c = __builtin_amdgcn_wmma_f32_16x16x32_bf16(false, a, false, b, (short)0, c, false, false);
  }
  return c;
}

// ---------------- in_proj GEMM: [32768,96] x [384,96]^T, LDS-staged ----------------
// Block: 256 thr (8 waves). Strip = 64 rows x 192 out-cols (half of 384).
// The n-half is block-uniform: half 0 -> xpart (NCHW), half 1 -> z. Scalar branch only.
__global__ void gemm_in(const __bf16* __restrict__ Abf, const __bf16* __restrict__ Wbf,
                        float* __restrict__ xpart_nchw, float* __restrict__ zraw) {
  __shared__ uint4 AsRaw[64 * DM * 2 / 16];    // 12 KB
  __shared__ uint4 BsRaw[192 * DM * 2 / 16];   // 36 KB
  __bf16* As = (__bf16*)AsRaw;
  __bf16* Bs = (__bf16*)BsRaw;

  int m0 = (blockIdx.x >> 1) * 64;
  int o0 = (blockIdx.x & 1) * 192;
  bool to_x = (o0 == 0);               // block-uniform predicate

  // coalesced 16B cooperative loads into LDS
  const uint4* gA = (const uint4*)(Abf + (size_t)m0 * DM);
  for (int i = threadIdx.x; i < 64 * DM * 2 / 16; i += 256) AsRaw[i] = gA[i];
  const uint4* gB = (const uint4*)(Wbf + (size_t)o0 * DM);
  for (int i = threadIdx.x; i < 192 * DM * 2 / 16; i += 256) BsRaw[i] = gB[i];
  __syncthreads();

  int wv = threadIdx.x >> 5, lane = threadIdx.x & 31;
  int grp = lane >> 4;
  // 4 m-subtiles x 12 n-subtiles = 48 tiles, 8 waves -> 6 tiles/wave (mt wave-invariant)
  for (int t = wv; t < 48; t += 8) {
    int mt = t & 3, nt = t >> 2;
    v8f c = wmma_tile<DM>(As + mt * 16 * DM, Bs + nt * 16 * DM, lane);
    int colh = nt * 16 + (lane & 15);           // 0..191 within the half
    if (to_x) {
      // xpart -> NCHW for depthwise conv (channel = colh)
#pragma unroll
      for (int r = 0; r < 8; r++) {
        int m = m0 + mt * 16 + r + grp * 8;
        int b = m >> 12, pos = m & (LL - 1);
        xpart_nchw[(((size_t)b * DIN + colh) << 12) + pos] = c[r];
      }
    } else {
      // z (raw; silu applied at gating) — coalesced across lanes
#pragma unroll
      for (int r = 0; r < 8; r++) {
        int m = m0 + mt * 16 + r + grp * 8;
        zraw[(size_t)m * DIN + colh] = c[r];
      }
    }
  }
}

// ---------------- out_proj GEMM: [32768,192] x [96,192]^T -> d_out, LDS-staged ----------------
__global__ void gemm_out(const __bf16* __restrict__ Abf, const __bf16* __restrict__ Wbf,
                         float* __restrict__ out) {
  __shared__ uint4 AsRaw[64 * DIN * 2 / 16];   // 24 KB
  __shared__ uint4 BsRaw[96 * DIN * 2 / 16];   // 36 KB
  __bf16* As = (__bf16*)AsRaw;
  __bf16* Bs = (__bf16*)BsRaw;

  int m0 = blockIdx.x * 64;

  const uint4* gA = (const uint4*)(Abf + (size_t)m0 * DIN);
  for (int i = threadIdx.x; i < 64 * DIN * 2 / 16; i += 256) AsRaw[i] = gA[i];
  const uint4* gB = (const uint4*)Wbf;
  for (int i = threadIdx.x; i < 96 * DIN * 2 / 16; i += 256) BsRaw[i] = gB[i];
  __syncthreads();

  int wv = threadIdx.x >> 5, lane = threadIdx.x & 31;
  int grp = lane >> 4;
  // 4 m-subtiles x 6 n-subtiles = 24 tiles, 8 waves -> 3 tiles/wave
  for (int t = wv; t < 24; t += 8) {
    int mt = t & 3, nt = t >> 2;
    v8f c = wmma_tile<DIN>(As + mt * 16 * DIN, Bs + nt * 16 * DIN, lane);
    int col = nt * 16 + (lane & 15);
#pragma unroll
    for (int r = 0; r < 8; r++) {
      int m = m0 + mt * 16 + r + grp * 8;
      out[(size_t)m * DM + col] = c[r];
    }
  }
}

// ---------------- depthwise 3x3 conv + bias + silu ----------------
__global__ void conv_dw_silu(const float* __restrict__ xp, const float* __restrict__ cw,
                             const float* __restrict__ cb, float* __restrict__ xc) {
  int idx = blockIdx.x * blockDim.x + threadIdx.x;
  if (idx >= BQ * DIN * LL) return;
  int w = idx & 63, h = (idx >> 6) & 63;
  int c = (idx >> 12) % DIN, b = idx / (DIN * LL);
  const float* base = xp + ((size_t)b * DIN + c) * LL;
  float acc = cb[c];
#pragma unroll
  for (int i = 0; i < 3; i++) {
    int hh = h + i - 1;
    if ((unsigned)hh >= 64u) continue;
#pragma unroll
    for (int j = 0; j < 3; j++) {
      int ww = w + j - 1;
      if ((unsigned)ww >= 64u) continue;
      acc += base[hh * 64 + ww] * cw[c * 9 + i * 3 + j];
    }
  }
  xc[idx] = acc / (1.0f + __expf(-acc));   // silu
}

// ---------------- x_proj: x_dbl[b,k,c,l] = sum_d xs[b,k,d,l] * x_proj_w[k,c,d] ----------------
__global__ void xproj(const float* __restrict__ xc, const float* __restrict__ xpw,
                      float* __restrict__ xdbl) {
  int idx = blockIdx.x * blockDim.x + threadIdx.x;
  if (idx >= BQ * KD * CDBL * LL) return;
  int l = idx & (LL - 1);
  int c = (idx >> 12) % CDBL;
  int k = (idx / (CDBL * LL)) % KD;
  int b = idx / (KD * CDBL * LL);
  int pos = perm_pos(k, l);
  const float* xcb = xc + (size_t)b * DIN * LL + pos;
  const float* wp  = xpw + (size_t)(k * CDBL + c) * DIN;
  float acc = 0.0f;
  for (int d = 0; d < DIN; d++) acc += xcb[(size_t)d * LL] * wp[d];
  xdbl[idx] = acc;
}

// ---------------- selective scan: one lane per (b,k,d); merged via f32 atomics ----------------
__global__ void scan_kernel(const float* __restrict__ xc, const float* __restrict__ xdbl,
                            const float* __restrict__ Alogs, const float* __restrict__ dtw,
                            const float* __restrict__ dtb, const float* __restrict__ Dsp,
                            float* __restrict__ ym) {
  int gid = blockIdx.x * blockDim.x + threadIdx.x;
  if (gid >= BQ * KD * DIN) return;
  int d  = gid % DIN;
  int bk = gid / DIN;      // wave-uniform: 192 = 6 waves per (b,k)
  int k = bk & 3, b = bk >> 2;
  int bk_u = __builtin_amdgcn_readfirstlane(bk);   // provably uniform -> scalar loads

  float A[NST], h[NST], dw[RNK];
#pragma unroll
  for (int n = 0; n < NST; n++) { A[n] = -__expf(Alogs[(size_t)(k * DIN + d) * NST + n]); h[n] = 0.0f; }
#pragma unroll
  for (int r = 0; r < RNK; r++) dw[r] = dtw[(size_t)(k * DIN + d) * RNK + r];
  float bias = dtb[k * DIN + d];
  float Dv   = Dsp[k * DIN + d];

  const float* xdb = xdbl + (size_t)bk_u * CDBL * LL;
  const float* xcb = xc + (size_t)b * DIN * LL + (size_t)d * LL;
  float*       yb  = ym + (size_t)b * DIN * LL + (size_t)d * LL;

  for (int l = 0; l < LL; l++) {
    if ((l & 31) == 0 && l + 32 < LL) {
      // prefetch next cacheline of every x_dbl row (global_prefetch_b8)
#pragma unroll
      for (int c = 0; c < CDBL; c++)
        __builtin_prefetch(xdb + (size_t)c * LL + l + 32, 0, 1);
    }
    int pos = perm_pos(k, l);
    float u = xcb[pos];
    float dtr = bias;
#pragma unroll
    for (int r = 0; r < RNK; r++) dtr += xdb[(size_t)r * LL + l] * dw[r];
    float delta = (dtr > 20.0f) ? dtr : log1pf(__expf(dtr));   // softplus
    float du = delta * u;
    float y = 0.0f;
#pragma unroll
    for (int n = 0; n < NST; n++) {
      float Bn = xdb[(size_t)(RNK + n) * LL + l];
      float Cn = xdb[(size_t)(RNK + NST + n) * LL + l];
      h[n] = __expf(delta * A[n]) * h[n] + du * Bn;
      y += h[n] * Cn;
    }
    atomicAdd(yb + pos, y + Dv * u);   // cross_merge scatter (global_atomic_add_f32)
  }
}

// ---------------- layernorm over d + silu(z) gate -> bf16 ----------------
__global__ void norm_gate(const float* __restrict__ ym, const float* __restrict__ zraw,
                          const float* __restrict__ nw, const float* __restrict__ nb,
                          __bf16* __restrict__ yg) {
  int t = blockIdx.x * blockDim.x + threadIdx.x;
  if (t >= MTOK) return;
  int b = t >> 12, l = t & (LL - 1);
  const float* yb = ym + (size_t)b * DIN * LL + l;
  float s = 0.0f, s2 = 0.0f;
  for (int d = 0; d < DIN; d++) { float v = yb[(size_t)d * LL]; s += v; s2 += v * v; }
  float mu = s * (1.0f / DIN);
  float var = s2 * (1.0f / DIN) - mu * mu;
  float rs = rsqrtf(var + 1e-5f);
  const float* zb = zraw + (size_t)t * DIN;
  __bf16* og = yg + (size_t)t * DIN;
  for (int d = 0; d < DIN; d++) {
    float v = (yb[(size_t)d * LL] - mu) * rs * nw[d] + nb[d];
    float z = zb[d];
    float g = z / (1.0f + __expf(-z));
    og[d] = (__bf16)(v * g);
  }
}

extern "C" void kernel_launch(void* const* d_in, const int* in_sizes, int n_in,
                              void* d_out, int out_size, void* d_ws, size_t ws_size,
                              hipStream_t stream) {
  const float* x        = (const float*)d_in[0];
  const float* in_w     = (const float*)d_in[1];
  const float* conv_w   = (const float*)d_in[2];
  const float* conv_b   = (const float*)d_in[3];
  const float* xproj_w  = (const float*)d_in[4];
  const float* dt_w     = (const float*)d_in[5];
  const float* dt_b     = (const float*)d_in[6];
  const float* A_logs   = (const float*)d_in[7];
  const float* Ds       = (const float*)d_in[8];
  const float* norm_w   = (const float*)d_in[9];
  const float* norm_b   = (const float*)d_in[10];
  const float* out_w    = (const float*)d_in[11];
  float* out = (float*)d_out;

  char* w = (char*)d_ws;
  size_t off = 0;
  auto alloc = [&](size_t bytes) -> void* {
    void* p = w + off;
    off = (off + bytes + 255) & ~(size_t)255;
    return p;
  };
  __bf16* xbf   = (__bf16*)alloc((size_t)MTOK * DM * 2);
  __bf16* wibf  = (__bf16*)alloc((size_t)2 * DIN * DM * 2);
  __bf16* wobf  = (__bf16*)alloc((size_t)DM * DIN * 2);
  float*  xpart = (float*)alloc((size_t)BQ * DIN * LL * 4);
  float*  zraw  = (float*)alloc((size_t)MTOK * DIN * 4);
  float*  xc    = (float*)alloc((size_t)BQ * DIN * LL * 4);
  float*  xdbl  = (float*)alloc((size_t)BQ * KD * CDBL * LL * 4);
  float*  ym    = (float*)alloc((size_t)BQ * DIN * LL * 4);
  __bf16* yg    = (__bf16*)alloc((size_t)MTOK * DIN * 2);

  // 1) pack bf16 operands
  pack_bf16<<<(MTOK * DM + 255) / 256, 256, 0, stream>>>(x, xbf, MTOK * DM);
  pack_bf16<<<(2 * DIN * DM + 255) / 256, 256, 0, stream>>>(in_w, wibf, 2 * DIN * DM);
  pack_bf16<<<(DM * DIN + 255) / 256, 256, 0, stream>>>(out_w, wobf, DM * DIN);

  // 2) in_proj (WMMA, LDS-staged): 512 m-strips x 2 n-halves
  gemm_in<<<512 * 2, 256, 0, stream>>>(xbf, wibf, xpart, zraw);

  // 3) depthwise conv + silu
  conv_dw_silu<<<(BQ * DIN * LL + 255) / 256, 256, 0, stream>>>(xpart, conv_w, conv_b, xc);

  // 4) zero merge buffer, x_proj
  zero_f32<<<(BQ * DIN * LL + 255) / 256, 256, 0, stream>>>(ym, BQ * DIN * LL);
  xproj<<<(BQ * KD * CDBL * LL + 255) / 256, 256, 0, stream>>>(xc, xproj_w, xdbl);

  // 5) selective scan + fused dt_proj + atomic cross_merge (6144 lanes)
  scan_kernel<<<(BQ * KD * DIN + 255) / 256, 256, 0, stream>>>(xc, xdbl, A_logs, dt_w, dt_b, Ds, ym);

  // 6) layernorm + gate -> bf16
  norm_gate<<<(MTOK + 255) / 256, 256, 0, stream>>>(ym, zraw, norm_w, norm_b, yg);

  // 7) out_proj (WMMA, LDS-staged): 512 m-strips
  gemm_out<<<512, 256, 0, stream>>>(yg, wobf, out);
}
